// TransducerJointNetwork_88141318848779
// MI455X (gfx1250) — matrix operand order repaired and checked by reading, not compiled, not run on hardware
//
#include <hip/hip_runtime.h>
#include <hip/hip_bf16.h>

typedef __attribute__((ext_vector_type(16))) _Float16 v16h;
typedef __attribute__((ext_vector_type(8)))  float    v8f;

#if defined(__HIP_DEVICE_COMPILE__) && __has_builtin(__builtin_amdgcn_tanhf)
#  define FAST_TANH(x) __builtin_amdgcn_tanhf(x)
#else
#  define FAST_TANH(x) tanhf(x)
#endif

// Problem dims (fixed by the reference)
#define B_  4
#define T_  256
#define U_  128
#define E_  512
#define D_  640
#define J_  640
#define V_  1024

// ---------------------------------------------------------------------------
// Small projection GEMM: P[row, j] = sum_k X[row,k]*W[k,j] + bias[j]  (f32)
// ~1.1 GFLOP total: negligible vs the joint GEMM.
// ---------------------------------------------------------------------------
__global__ __launch_bounds__(256)
void proj_kernel(const float* __restrict__ X, const float* __restrict__ W,
                 const float* __restrict__ bias, float* __restrict__ P,
                 int K, int J)
{
    __shared__ float xrow[D_]; // max K = 640
    const int row = blockIdx.x;
    for (int k = threadIdx.x; k < K; k += blockDim.x)
        xrow[k] = X[(size_t)row * K + k];
    __syncthreads();
    for (int j = threadIdx.x; j < J; j += blockDim.x) {
        float acc = bias[j];
#pragma unroll 4
        for (int k = 0; k < K; ++k)
            acc = fmaf(xrow[k], W[(size_t)k * J + j], acc);
        P[(size_t)row * J + j] = acc;
    }
}

// ---------------------------------------------------------------------------
// Pack W_joint (J x V, f32, row-major) into f16 B-fragments for
// v_wmma_f32_16x16x32_f16. Tile = 32(K) x 16(N). ISA B layout:
// lane L<16 -> col N=L, K=0..7,16..23; lane L>=16 -> col N=L-16,
// K=8..15,24..31. Storage: Wp[((kt*(V/16)+vt)*32 + lane)*16 + i] -> each lane's
// 16 halves contiguous (two b128 loads at consume time).
// ---------------------------------------------------------------------------
__global__ __launch_bounds__(256)
void wjoint_pack_kernel(const float* __restrict__ Wj, _Float16* __restrict__ Wp)
{
    const int g = blockIdx.x * blockDim.x + threadIdx.x;
    if (g >= (J_ / 32) * (V_ / 16) * 32) return;
    const int lane = g & 31;
    const int tile = g >> 5;
    const int vt   = tile & (V_ / 16 - 1);
    const int kt   = tile / (V_ / 16);
    const int n    = vt * 16 + (lane & 15);
    const int kb   = kt * 32 + ((lane & 16) ? 8 : 0);
    _Float16* dst = Wp + (size_t)g * 16;
#pragma unroll
    for (int i = 0; i < 16; ++i)
        dst[i] = (_Float16)Wj[(size_t)(kb + i + (i >= 8 ? 8 : 0)) * V_ + n];
}

// ---------------------------------------------------------------------------
// Fused joint kernel:
//   out[b,t,u,v] = tanh(encP[b,t,:] + decP[b,u,:]) @ W_joint[:,v] + b_joint[v]
//
// Block = 256 threads = 8 waves. Block tile = 128 rows (all U for one (b,t))
// x 256 cols. Grid = (B*T, V/256). Wave (wr = w>>1, wc = w&1) computes a
// 32-row x 128-col sub-tile: 2 A fragments x 8 B fragments -> 16 WMMAs and
// 16 f32 accumulators per K step. A tile (128x32 f16) is computed (tanh of
// broadcast sum) into double-buffered LDS each K step; one barrier per iter.
// B is read from the pre-swizzled, L2-resident Wp (amortized over 128 rows).
// ---------------------------------------------------------------------------
__global__ __launch_bounds__(256)
void joint_wmma_kernel(const float* __restrict__ encP,
                       const float* __restrict__ decP,
                       const _Float16* __restrict__ Wp,
                       const float* __restrict__ bj,
                       float* __restrict__ out)
{
    // 8 row-fragments of 16x32 f16, double buffered: 16 KB
    __shared__ _Float16 aBuf[2][8][512];

    const int tid  = threadIdx.x;
    const int lane = tid & 31;
    const int w    = tid >> 5;

    // Block decode
    const int bt = blockIdx.x;          // b*T + t
    const int b  = bt >> 8;             // /T_
    const float* encRow  = encP + (size_t)bt * J_;
    const float* decRows = decP + (size_t)(b * U_) * J_;   // all 128 u-rows

    // --- A staging mapping: each thread produces one lane-fragment (16 halves)
    // Fragment (sTile, sLane): row M = sTile*16 + sLane%16.
    // ISA A octet layout: half i -> K = i + 8*[i>=8] + 8*[sLane>=16], i.e. two
    // runs of 8 consecutive K: [off, off+8) and [16+off, 16+off+8).
    const int sTile = tid >> 5;                 // 0..7
    const int sLane = lane;
    const int sRow  = sTile * 16 + (sLane & 15);      // u in 0..127
    const int sOff  = (sLane & 16) ? 8 : 0;
    const float* ePtr = encRow + sOff;
    const float* dPtr = decRows + (size_t)sRow * J_ + sOff;

    // --- Wave tiling: rows wr*32 + {0,16}; cols blockIdx.y*256 + wc*128
    const int wr    = w >> 1;                   // 0..3
    const int wc    = w & 1;                    // 0..1
    const int vbase = blockIdx.y * 256 + wc * 128;
    const int vt0   = vbase >> 4;

    v8f c[2][8];
#pragma unroll
    for (int m = 0; m < 2; ++m)
#pragma unroll
        for (int n = 0; n < 8; ++n) c[m][n] = (v8f){};

    for (int kk = 0; kk < J_ / 32; ++kk) {
        const int K0 = kk * 32;
        const int p  = kk & 1;

        // ---- Stage A: 4x float4 from enc + 4x float4 from dec, 16 tanh,
        //      one 32B LDS store in fragment layout.
        {
            float ev[16], dv[16];
            *(float4*)&ev[0]  = *(const float4*)(ePtr + K0);
            *(float4*)&ev[4]  = *(const float4*)(ePtr + K0 + 4);
            *(float4*)&ev[8]  = *(const float4*)(ePtr + K0 + 16);
            *(float4*)&ev[12] = *(const float4*)(ePtr + K0 + 20);
            *(float4*)&dv[0]  = *(const float4*)(dPtr + K0);
            *(float4*)&dv[4]  = *(const float4*)(dPtr + K0 + 4);
            *(float4*)&dv[8]  = *(const float4*)(dPtr + K0 + 16);
            *(float4*)&dv[12] = *(const float4*)(dPtr + K0 + 20);
            v16h hv;
#pragma unroll
            for (int i = 0; i < 16; ++i)
                hv[i] = (_Float16)FAST_TANH(ev[i] + dv[i]);
            *(v16h*)&aBuf[p][sTile][sLane * 16] = hv;
        }
        __syncthreads();

        // ---- Consume: 2 A fragments (LDS), 8 B fragments (L2), 16 WMMAs
        v16h a0 = *(const v16h*)&aBuf[p][wr * 2 + 0][lane * 16];
        v16h a1 = *(const v16h*)&aBuf[p][wr * 2 + 1][lane * 16];
        const _Float16* wp0 =
            Wp + ((size_t)(kk * (V_ / 16) + vt0) * 32 + lane) * 16;
#pragma unroll
        for (int n = 0; n < 8; ++n) {
            v16h bfrag = *(const v16h*)(wp0 + (size_t)n * 512);
            c[0][n] = __builtin_amdgcn_wmma_f32_16x16x32_f16(
                false, a0, false, bfrag, (short)0, c[0][n], false, false);
            c[1][n] = __builtin_amdgcn_wmma_f32_16x16x32_f16(
                false, a1, false, bfrag, (short)0, c[1][n], false, false);
        }
        // Double buffering: single barrier per iteration is sufficient.
    }

    // ---- Epilogue: C layout: VGPR r -> row r (lanes 0-15) / row r+8
    // (lanes 16-31); col = lane%16. Add bias, stream f32 stores (512 MB total).
    const int colN     = lane & 15;
    const int rowBase0 = bt * U_ + wr * 32 + ((lane & 16) ? 8 : 0);
#pragma unroll
    for (int n = 0; n < 8; ++n) {
        const int col = vbase + n * 16 + colN;
        const float bias = bj[col];
#pragma unroll
        for (int r = 0; r < 8; ++r) {
            out[(size_t)(rowBase0 + r) * V_ + col]      = c[0][n][r] + bias;
            out[(size_t)(rowBase0 + 16 + r) * V_ + col] = c[1][n][r] + bias;
        }
    }
}

// ---------------------------------------------------------------------------
extern "C" void kernel_launch(void* const* d_in, const int* in_sizes, int n_in,
                              void* d_out, int out_size, void* d_ws, size_t ws_size,
                              hipStream_t stream)
{
    const float* enc     = (const float*)d_in[0];
    const float* dec     = (const float*)d_in[1];
    const float* W_enc   = (const float*)d_in[2];
    const float* b_enc   = (const float*)d_in[3];
    const float* W_dec   = (const float*)d_in[4];
    const float* b_dec   = (const float*)d_in[5];
    const float* W_joint = (const float*)d_in[6];
    const float* b_joint = (const float*)d_in[7];
    float* out = (float*)d_out;

    // Workspace layout (5 MB total, all L2-resident):
    float*    encP = (float*)d_ws;                               // (B*T, J) f32
    float*    decP = encP + (size_t)(B_ * T_) * J_;              // (B*U, J) f32
    _Float16* Wp   = (_Float16*)(decP + (size_t)(B_ * U_) * J_); // packed f16

    // 1) Projections (+bias) into workspace
    proj_kernel<<<B_ * T_, 256, 0, stream>>>(enc, W_enc, b_enc, encP, E_, J_);
    proj_kernel<<<B_ * U_, 256, 0, stream>>>(dec, W_dec, b_dec, decP, D_, J_);

    // 2) Pack W_joint into WMMA B-fragment layout (f16)
    const int packThreads = (J_ / 32) * (V_ / 16) * 32;          // 40960
    wjoint_pack_kernel<<<(packThreads + 255) / 256, 256, 0, stream>>>(W_joint, Wp);

    // 3) Fused tanh-join + GEMM + bias -> logits
    dim3 grid(B_ * T_, V_ / 256);
    joint_wmma_kernel<<<grid, 256, 0, stream>>>(encP, decP, Wp, b_joint, out);
}
